// GNN_57294863728768
// MI455X (gfx1250) — matrix operand order repaired and checked by reading, not compiled, not run on hardware
//
#include <hip/hip_runtime.h>

typedef unsigned short u16;
typedef __attribute__((ext_vector_type(8)))  u16    u16x8;
typedef __attribute__((ext_vector_type(16))) u16    u16x16;
typedef __attribute__((ext_vector_type(16))) __bf16 v16bf;
typedef __attribute__((ext_vector_type(8)))  float  v8f;

#define NN 50000
#define EE 400000

// ---------- helpers ----------
__device__ __forceinline__ u16 f2bf(float f) {
    unsigned u = __builtin_bit_cast(unsigned, f);
    u += 0x7FFFu + ((u >> 16) & 1u);
    return (u16)(u >> 16);
}
__device__ __forceinline__ float bf2f(u16 h) {
    unsigned u = ((unsigned)h) << 16;
    return __builtin_bit_cast(float, u);
}
__device__ __forceinline__ v16bf mkfrag(u16x8 lo, u16x8 hi) {
    u16x16 c = __builtin_shufflevector(lo, hi, 0,1,2,3,4,5,6,7,8,9,10,11,12,13,14,15);
    return __builtin_bit_cast(v16bf, c);
}
__device__ __forceinline__ v16bf ldfrag(const u16* p) {
    u16x16 c = *(const u16x16*)p;
    return __builtin_bit_cast(v16bf, c);
}
__device__ __forceinline__ v8f wmma_bf16(v16bf a, v16bf b, v8f c) {
    return __builtin_amdgcn_wmma_f32_16x16x32_bf16(false, a, false, b, (short)0, c, false, false);
}

// ---------- tiny utility kernels ----------
__global__ void zero_kernel(float* p, int n) {
    int i = blockIdx.x * blockDim.x + threadIdx.x;
    if (i < n) p[i] = 0.f;
}
__global__ void f2bf_kernel(const float* __restrict__ x, u16* __restrict__ y, int n) {
    int i = blockIdx.x * blockDim.x + threadIdx.x;
    if (i < n) y[i] = f2bf(x[i]);
}
// pack f32 weights [K,Nc] row-major into WMMA B-fragment layout (bf16):
// out[((kt*ntiles+nt)*32 + lane)*16 + i] = W[(kt*32 + 8*(lane>>4) + (i&7) + 16*(i>>3))*Nc + nt*16 + (lane&15)]
__global__ void pack_b_kernel(const float* __restrict__ W, u16* __restrict__ out, int K, int Nc) {
    int o = blockIdx.x * blockDim.x + threadIdx.x;
    if (o >= K * Nc) return;
    int i    = o & 15;
    int lane = (o >> 4) & 31;
    int tidx = o >> 9;
    int ntiles = Nc >> 4;
    int nt = tidx % ntiles;
    int kt = tidx / ntiles;
    int c = nt * 16 + (lane & 15);
    int k = kt * 32 + 8 * (lane >> 4) + (i & 7) + 16 * (i >> 3);
    out[o] = f2bf(W[k * Nc + c]);
}
// edge degrees (hop-invariant)
__global__ void deg_kernel(const int* __restrict__ d1, const int* __restrict__ d0,
                           float* __restrict__ deg1, float* __restrict__ deg0, int E) {
    int e = blockIdx.x * blockDim.x + threadIdx.x;
    if (e < E) {
        unsafeAtomicAdd(&deg1[d1[e]], 1.f);
        unsafeAtomicAdd(&deg0[d0[e]], 1.f);
    }
}
// per-label constant: c[j] = b1x[j] + sum_k he[k] * W1x[(256+k)*256 + j],
// where he = MLP_e(scalar label). One block of 256 threads.
__global__ void edge_const_kernel(const float* __restrict__ lab,
                                  const float* __restrict__ W1e, const float* __restrict__ b1e,
                                  const float* __restrict__ W2e, const float* __restrict__ b2e,
                                  const float* __restrict__ W3e, const float* __restrict__ b3e,
                                  const float* __restrict__ W1x, const float* __restrict__ b1x,
                                  float* __restrict__ cvec) {
    __shared__ float sh1[256];
    __shared__ float sh2[128];
    __shared__ float she[128];
    int t = threadIdx.x;
    float x = lab[0];
    sh1[t] = fmaxf(W1e[t] * x + b1e[t], 0.f);
    __syncthreads();
    if (t < 128) {
        float a = b2e[t];
        for (int k = 0; k < 256; ++k) a += sh1[k] * W2e[k * 128 + t];
        sh2[t] = fmaxf(a, 0.f);
    }
    __syncthreads();
    if (t < 128) {
        float a = b3e[t];
        for (int k = 0; k < 128; ++k) a += sh2[k] * W3e[k * 128 + t];
        she[t] = a;
    }
    __syncthreads();
    float a = b1x[t];
    for (int k = 0; k < 128; ++k) a += she[k] * W1x[(256 + k) * 256 + t];
    cvec[t] = a;
}
// segment mean + build concat [hb | avg_m1 | avg_m0] as bf16
__global__ void segmean_concat_kernel(const u16* __restrict__ hb,
                                      const float* __restrict__ sum1, const float* __restrict__ deg1,
                                      const float* __restrict__ sum0, const float* __restrict__ deg0,
                                      u16* __restrict__ cat, int n) {
    int i = blockIdx.x * blockDim.x + threadIdx.x;
    if (i >= n) return;
    int node = i >> 7, c = i & 127;
    float a1 = sum1[i] / fmaxf(deg1[node], 1.f);
    float a0 = sum0[i] / fmaxf(deg0[node], 1.f);
    u16* row = cat + (size_t)node * 384;
    row[c]       = hb[i];
    row[128 + c] = f2bf(a1);
    row[256 + c] = f2bf(a0);
}

// ---------- generic WMMA GEMM: out = [relu](A[M,K]bf16 @ Bpacked[K,Nc] + bias) (+resid) ----------
__global__ __launch_bounds__(128) void gemm_wmma_kernel(
        const u16* __restrict__ A, const u16* __restrict__ Bp, const float* __restrict__ bias,
        int M, int K, int Nc, int do_relu,
        u16* __restrict__ outb, float* __restrict__ outf, const float* __restrict__ resid) {
    int lane = threadIdx.x & 31;
    int wv   = threadIdx.x >> 5;
    int ntiles = Nc >> 4;
    int tile = blockIdx.x * 4 + wv;
    if (tile >= (M >> 4) * ntiles) return;            // uniform per wave: EXEC stays full
    int mt = tile / ntiles, nt = tile % ntiles;
    int hf = lane >> 4, r = lane & 15;
    int ktiles = K >> 5;
    float bv = bias ? bias[nt * 16 + r] : 0.f;
    v8f acc;
    #pragma unroll
    for (int j = 0; j < 8; ++j) acc[j] = bv;
    const u16* Arow = A + (size_t)(mt * 16 + r) * K;
    for (int kt = 0; kt < ktiles; ++kt) {
        u16x8 a0 = *(const u16x8*)(Arow + kt * 32 + 8 * hf);
        u16x8 a1 = *(const u16x8*)(Arow + kt * 32 + 16 + 8 * hf);
        v16bf af = mkfrag(a0, a1);
        v16bf bf = ldfrag(Bp + ((size_t)(kt * ntiles + nt) * 32 + lane) * 16);
        acc = wmma_bf16(af, bf, acc);
    }
    #pragma unroll
    for (int j = 0; j < 8; ++j) {
        int row = mt * 16 + j + 8 * hf;
        int col = nt * 16 + r;
        float v = acc[j];
        if (do_relu) v = fmaxf(v, 0.f);
        size_t o = (size_t)row * Nc + col;
        if (resid) v += resid[o];
        if (outf) outf[o] = v;
        if (outb) outb[o] = f2bf(v);
    }
}

// ---------- fused per-edge MLP (layers 2+3) with layer-1 gather-add-relu ----------
// Per wave: 16 edges. z1[16,256] = relu(Psrc[src] + Pdst[dst] + c) built directly as A-frags.
// z2 = relu(z1 @ W2 + b2) -> staged through LDS for transpose -> m = z2 @ W3 + b3
// -> atomic scatter-add into sum[dst].
__global__ __launch_bounds__(256) void edge_mlp_kernel(
        const u16* __restrict__ Psrc, const u16* __restrict__ Pdst,
        const float* __restrict__ cvec,
        const u16* __restrict__ W2p, const u16* __restrict__ W3p,
        const float* __restrict__ b2, const float* __restrict__ b3,
        const int* __restrict__ src, const int* __restrict__ dst,
        float* __restrict__ sum) {
    __shared__ u16 sW3[128 * 128];      // 32 KB: packed W3 fragments
    __shared__ u16 stage[8 * 16 * 128]; // 32 KB: per-wave z2 transpose staging
    {   // cooperative copy of packed W3 (16384 u16) into LDS
        const u16x8* s8 = (const u16x8*)W3p;
        u16x8* d8 = (u16x8*)sW3;
        #pragma unroll
        for (int i = 0; i < 8; ++i) d8[threadIdx.x + i * 256] = s8[threadIdx.x + i * 256];
    }
    __syncthreads();
    int wave = threadIdx.x >> 5;
    int lane = threadIdx.x & 31;
    int hf = lane >> 4, r = lane & 15;
    int e0 = (blockIdx.x * 8 + wave) * 16;
    int s = src[e0 + r];
    int d = dst[e0 + r];
    int dj[8];
    #pragma unroll
    for (int j = 0; j < 8; ++j) dj[j] = dst[e0 + j + 8 * hf];
    const u16* prs = Psrc + (size_t)s * 256;
    const u16* prd = Pdst + (size_t)d * 256;

    // ---- layer 2: [16,256] @ [256,128] ----
    v8f acc[8];
    #pragma unroll
    for (int nt = 0; nt < 8; ++nt) {
        float bv = b2[nt * 16 + r];
        #pragma unroll
        for (int j = 0; j < 8; ++j) acc[nt][j] = bv;
    }
    #pragma unroll
    for (int kt = 0; kt < 8; ++kt) {
        int c0 = kt * 32 + 8 * hf;
        int c1 = c0 + 16;
        u16x8 ps0 = *(const u16x8*)(prs + c0);
        u16x8 pd0 = *(const u16x8*)(prd + c0);
        u16x8 ps1 = *(const u16x8*)(prs + c1);
        u16x8 pd1 = *(const u16x8*)(prd + c1);
        u16x8 a0, a1;
        #pragma unroll
        for (int j = 0; j < 8; ++j) {
            float v0 = bf2f(ps0[j]) + bf2f(pd0[j]) + cvec[c0 + j];
            float v1 = bf2f(ps1[j]) + bf2f(pd1[j]) + cvec[c1 + j];
            a0[j] = f2bf(fmaxf(v0, 0.f));
            a1[j] = f2bf(fmaxf(v1, 0.f));
        }
        v16bf af = mkfrag(a0, a1);
        #pragma unroll
        for (int nt = 0; nt < 8; ++nt) {
            v16bf bf = ldfrag(W2p + ((size_t)(kt * 8 + nt) * 32 + lane) * 16);
            acc[nt] = wmma_bf16(af, bf, acc[nt]);
        }
    }
    // ---- relu + stage z2 to LDS (wave-private transpose) ----
    u16* myst = stage + wave * 2048;
    #pragma unroll
    for (int nt = 0; nt < 8; ++nt)
        #pragma unroll
        for (int j = 0; j < 8; ++j)
            myst[(j + 8 * hf) * 128 + nt * 16 + r] = f2bf(fmaxf(acc[nt][j], 0.f));

    // ---- layer 3: [16,128] @ [128,128] ----
    v8f acc3[8];
    #pragma unroll
    for (int nt = 0; nt < 8; ++nt) {
        float bv = b3[nt * 16 + r];
        #pragma unroll
        for (int j = 0; j < 8; ++j) acc3[nt][j] = bv;
    }
    const u16* zrow = myst + r * 128;
    #pragma unroll
    for (int kt = 0; kt < 4; ++kt) {
        u16x8 a0 = *(const u16x8*)(zrow + kt * 32 + 8 * hf);
        u16x8 a1 = *(const u16x8*)(zrow + kt * 32 + 16 + 8 * hf);
        v16bf af = mkfrag(a0, a1);
        #pragma unroll
        for (int nt = 0; nt < 8; ++nt) {
            v16bf bf = ldfrag(sW3 + ((kt * 8 + nt) * 32 + lane) * 16);
            acc3[nt] = wmma_bf16(af, bf, acc3[nt]);
        }
    }
    // ---- scatter-add into segment sums ----
    #pragma unroll
    for (int nt = 0; nt < 8; ++nt)
        #pragma unroll
        for (int j = 0; j < 8; ++j)
            unsafeAtomicAdd(&sum[(size_t)dj[j] * 128 + nt * 16 + r], acc3[nt][j]);
}

// ---------- host ----------
extern "C" void kernel_launch(void* const* d_in, const int* in_sizes, int n_in,
                              void* d_out, int out_size, void* d_ws, size_t ws_size,
                              hipStream_t stream) {
    const float* xv  = (const float*)d_in[0];
    const float* le1 = (const float*)d_in[1];
    const float* le0 = (const float*)d_in[2];
    const int* e1s = (const int*)d_in[3];
    const int* e1d = (const int*)d_in[4];
    const int* e0s = (const int*)d_in[5];
    const int* e0d = (const int*)d_in[6];
    const float* P[30];
    for (int i = 0; i < 30; ++i) P[i] = (const float*)d_in[7 + i];
    // params_v: P[0..5], params_e: P[6..11], params_e1: P[12..17], params_e0: P[18..23], params_aggr: P[24..29]

    char* w = (char*)d_ws;
    size_t off = 0;
    auto alloc = [&](size_t bytes) -> void* {
        void* p = w + off;
        off = (off + bytes + 255) & ~(size_t)255;
        return p;
    };
    u16*   xb   = (u16*)  alloc((size_t)NN * 128 * 2);
    u16*   t1b  = (u16*)  alloc((size_t)NN * 256 * 2);   // also aggr z1
    u16*   t2b  = (u16*)  alloc((size_t)NN * 128 * 2);   // also aggr z2
    float* hA   = (float*)alloc((size_t)NN * 128 * 4);
    float* hB   = (float*)alloc((size_t)NN * 128 * 4);
    u16*   hb   = (u16*)  alloc((size_t)NN * 128 * 2);
    u16*   Ps1  = (u16*)  alloc((size_t)NN * 256 * 2);
    u16*   Pd1  = (u16*)  alloc((size_t)NN * 256 * 2);
    u16*   Ps0  = (u16*)  alloc((size_t)NN * 256 * 2);
    u16*   Pd0  = (u16*)  alloc((size_t)NN * 256 * 2);
    float* sum1 = (float*)alloc((size_t)NN * 128 * 4);
    float* sum0 = (float*)alloc((size_t)NN * 128 * 4);
    float* deg1 = (float*)alloc((size_t)NN * 4);
    float* deg0 = (float*)alloc((size_t)NN * 4);
    u16*   cat  = (u16*)  alloc((size_t)NN * 384 * 2);
    float* c1   = (float*)alloc(256 * 4);
    float* c0   = (float*)alloc(256 * 4);
    // packed weights (bf16 B-fragment layout)
    u16* pvW1 = (u16*)alloc(128 * 256 * 2);
    u16* pvW2 = (u16*)alloc(256 * 128 * 2);
    u16* pvW3 = (u16*)alloc(128 * 128 * 2);
    u16* p1a  = (u16*)alloc(128 * 256 * 2);
    u16* p1b  = (u16*)alloc(128 * 256 * 2);
    u16* p1W2 = (u16*)alloc(256 * 128 * 2);
    u16* p1W3 = (u16*)alloc(128 * 128 * 2);
    u16* p0a  = (u16*)alloc(128 * 256 * 2);
    u16* p0b  = (u16*)alloc(128 * 256 * 2);
    u16* p0W2 = (u16*)alloc(256 * 128 * 2);
    u16* p0W3 = (u16*)alloc(128 * 128 * 2);
    u16* paW1 = (u16*)alloc(384 * 256 * 2);
    u16* paW2 = (u16*)alloc(256 * 128 * 2);
    u16* paW3 = (u16*)alloc(128 * 128 * 2);

    auto pack = [&](const float* W, u16* out, int K, int Nc) {
        int total = K * Nc;
        pack_b_kernel<<<(total + 255) / 256, 256, 0, stream>>>(W, out, K, Nc);
    };
    auto zerof = [&](float* p, int n) {
        zero_kernel<<<(n + 255) / 256, 256, 0, stream>>>(p, n);
    };
    auto gemm = [&](const u16* A, const u16* Bp, const float* bias, int M, int K, int Nc,
                    int relu, u16* outb, float* outf, const float* resid) {
        int tiles = (M / 16) * (Nc / 16);
        gemm_wmma_kernel<<<(tiles + 3) / 4, 128, 0, stream>>>(A, Bp, bias, M, K, Nc, relu,
                                                              outb, outf, resid);
    };

    // ---- one-time prep ----
    f2bf_kernel<<<(NN * 128 + 255) / 256, 256, 0, stream>>>(xv, xb, NN * 128);
    pack(P[0], pvW1, 128, 256); pack(P[2], pvW2, 256, 128); pack(P[4], pvW3, 128, 128);
    pack(P[12],            p1a, 128, 256); pack(P[12] + 128 * 256, p1b, 128, 256);
    pack(P[14], p1W2, 256, 128); pack(P[16], p1W3, 128, 128);
    pack(P[18],            p0a, 128, 256); pack(P[18] + 128 * 256, p0b, 128, 256);
    pack(P[20], p0W2, 256, 128); pack(P[22], p0W3, 128, 128);
    pack(P[24], paW1, 384, 256); pack(P[26], paW2, 256, 128); pack(P[28], paW3, 128, 128);
    edge_const_kernel<<<1, 256, 0, stream>>>(le1, P[6], P[7], P[8], P[9], P[10], P[11],
                                             P[12], P[13], c1);
    edge_const_kernel<<<1, 256, 0, stream>>>(le0, P[6], P[7], P[8], P[9], P[10], P[11],
                                             P[18], P[19], c0);
    zerof(deg1, NN); zerof(deg0, NN);
    deg_kernel<<<(EE + 255) / 256, 256, 0, stream>>>(e1d, e0d, deg1, deg0, EE);

    // ---- initial node MLP: h = MLP_v(x_v) ----
    gemm(xb,  pvW1, P[1], NN, 128, 256, 1, t1b, nullptr, nullptr);
    gemm(t1b, pvW2, P[3], NN, 256, 128, 1, t2b, nullptr, nullptr);
    gemm(t2b, pvW3, P[5], NN, 128, 128, 0, hb, hA, nullptr);

    float* hcur = hA;
    float* hnext = hB;
    for (int t = 0; t < 3; ++t) {
        zerof(sum1, NN * 128);
        zerof(sum0, NN * 128);
        // layer-1 node factorization per label
        gemm(hb, p1a, nullptr, NN, 128, 256, 0, Ps1, nullptr, nullptr);
        gemm(hb, p1b, nullptr, NN, 128, 256, 0, Pd1, nullptr, nullptr);
        gemm(hb, p0a, nullptr, NN, 128, 256, 0, Ps0, nullptr, nullptr);
        gemm(hb, p0b, nullptr, NN, 128, 256, 0, Pd0, nullptr, nullptr);
        // fused per-edge MLP + scatter-add
        edge_mlp_kernel<<<EE / 128, 256, 0, stream>>>(Ps1, Pd1, c1, p1W2, p1W3,
                                                      P[15], P[17], e1s, e1d, sum1);
        edge_mlp_kernel<<<EE / 128, 256, 0, stream>>>(Ps0, Pd0, c0, p0W2, p0W3,
                                                      P[21], P[23], e0s, e0d, sum0);
        // segment mean + concat
        segmean_concat_kernel<<<(NN * 128 + 255) / 256, 256, 0, stream>>>(
            hb, sum1, deg1, sum0, deg0, cat, NN * 128);
        // aggregation MLP with residual
        gemm(cat, paW1, P[25], NN, 384, 256, 1, t1b, nullptr, nullptr);
        gemm(t1b, paW2, P[27], NN, 256, 128, 1, t2b, nullptr, nullptr);
        float* outf = (t == 2) ? (float*)d_out : hnext;
        gemm(t2b, paW3, P[29], NN, 128, 128, 0, hb, outf, hcur);
        if (t < 2) { float* tmp = hcur; hcur = hnext; hnext = tmp; }
    }
}
